// Bmm1Strided_28724741275733
// MI455X (gfx1250) — compile-verified, matrix operand order checked by reading, not run on hardware
//
#include <hip/hip_runtime.h>

typedef __attribute__((ext_vector_type(16))) _Float16 v16h;
typedef __attribute__((ext_vector_type(8)))  _Float16 v8h;
typedef __attribute__((ext_vector_type(8)))  float    v8f;

#define HEADS     16
#define EMBED     64
#define QKV3      3
#define TOKSTRIDE (HEADS * QKV3 * EMBED) /* 3072 halves per token */
#define BATCH_N   64
#define MAXSEQ    512

// ---------------------------------------------------------------------------
// Kernel 1: prefix sums over seqlen -> token offsets + flat output offsets.
// seqlen is device data and we are under graph capture, so this runs on GPU.
// ---------------------------------------------------------------------------
__global__ void bmm1_prefix_kernel(const int* __restrict__ seqlen,
                                   long long* __restrict__ tok_offs,
                                   long long* __restrict__ out_offs,
                                   int batch) {
  if (blockIdx.x == 0 && threadIdx.x == 0) {
    long long t = 0, o = 0;
    for (int b = 0; b < batch; ++b) {
      tok_offs[b] = t;
      out_offs[b] = o;
      long long s = (long long)seqlen[b];
      t += s;
      o += (long long)HEADS * s * s;
    }
    tok_offs[batch] = t;
    out_offs[batch] = o;
  }
}

// ---------------------------------------------------------------------------
// Store one 16x16 f32 tile. s0/t0/S are SGPR-resident (s0, S from scalar
// sources; t0 laundered through readfirstlane), so the interior fast path is
// a single s_cbranch: one address setup + 8 NT stores with a running row
// pointer. Only true edge tiles pay per-element predication.
// ---------------------------------------------------------------------------
__device__ __forceinline__ void store_tile(float* __restrict__ out,
                                           long long obase, int S,
                                           int s0, int t0, int lane,
                                           const v8f& c) {
  const int n    = lane & 15;
  const int mhi  = (lane >> 4) * 8;
  const int tcol = t0 + n;
  if ((s0 + 16 <= S) && (t0 + 16 <= S)) {          // scalar (SGPR) fast path
    float* p = out + obase + (long long)(s0 + mhi) * S + tcol;
#pragma unroll
    for (int rr = 0; rr < 8; ++rr) {
      __builtin_nontemporal_store(c[rr] * 0.125f, p);
      p += S;
    }
  } else if (tcol < S) {                           // edge tile: masked
#pragma unroll
    for (int rr = 0; rr < 8; ++rr) {
      const int m = s0 + mhi + rr;
      if (m < S)
        __builtin_nontemporal_store(c[rr] * 0.125f,
                                    out + obase + (long long)m * S + tcol);
    }
  }
}

// ---------------------------------------------------------------------------
// Kernel 2: one wave computes a 16x32 output strip (two 16x16 t-tiles sharing
// one Q fragment): 4 x v_wmma_f32_16x16x32_f16, 12 x global_load_b128.
// All branches guarding WMMA are SGPR-based -> EXEC stays all-1s (ISA req).
// ---------------------------------------------------------------------------
__launch_bounds__(128)
__global__ void bmm1_wmma_kernel(const _Float16* __restrict__ mixed,
                                 const int* __restrict__ seqlen,
                                 const long long* __restrict__ tok_offs,
                                 const long long* __restrict__ out_offs,
                                 float* __restrict__ out) {
  const int bh = blockIdx.z;
  const int b  = bh >> 4;   // / HEADS
  const int h  = bh & 15;   // % HEADS
  const int S  = seqlen[b];

  const int s0 = blockIdx.y * 16;
  if (s0 >= S) return;                         // block-uniform exit

  // wave index is uniform across the wave but derived from threadIdx; launder
  // it into an SGPR so every guard below is a scalar branch (full EXEC).
  const int wave = __builtin_amdgcn_readfirstlane(threadIdx.x >> 5);
  const int lane = threadIdx.x & 31;
  const int t0   = (blockIdx.x * 4 + wave) * 32;  // 32-wide strip per wave
  if (t0 >= S) return;                            // s_cbranch exit
  const int t1 = t0 + 16;

  const long long tok0  = tok_offs[b];
  const long long obase = out_offs[b] + (long long)h * S * S;

  // WMMA 16-bit A/B fragment striping (wave32):
  //   lane L (L<16)  : row = L,    K-halves {kg..kg+7} & {kg+16..kg+23}, kg=0
  //   lane L (L>=16) : row = L-16, same chunks with kg=8
  const int row = lane & 15;
  const int kg  = (lane >> 4) ? 8 : 0;

  int srow  = s0 + row; if (srow >= S) srow = S - 1;   // clamp: EXEC stays full
  int trow0 = t0 + row; if (trow0 >= S) trow0 = S - 1;
  int trow1 = t1 + row; if (trow1 >= S) trow1 = S - 1;

  const long long hoff = h * (QKV3 * EMBED);
  const _Float16* qp  = mixed + (tok0 + srow)  * (long long)TOKSTRIDE + hoff + kg;
  const _Float16* kp0 = mixed + (tok0 + trow0) * (long long)TOKSTRIDE + hoff + EMBED + kg;
  const _Float16* kp1 = mixed + (tok0 + trow1) * (long long)TOKSTRIDE + hoff + EMBED + kg;

  // A fragment (shared by both t-tiles): four b128 loads.
  v8h a00 = *(const v8h*)(qp +  0);
  v8h a01 = *(const v8h*)(qp + 16);
  v8h a10 = *(const v8h*)(qp + 32);
  v8h a11 = *(const v8h*)(qp + 48);
  // B fragment, tile 0.
  v8h b00 = *(const v8h*)(kp0 +  0);
  v8h b01 = *(const v8h*)(kp0 + 16);
  v8h b10 = *(const v8h*)(kp0 + 32);
  v8h b11 = *(const v8h*)(kp0 + 48);

  v16h A0 = __builtin_shufflevector(a00, a01, 0,1,2,3,4,5,6,7,8,9,10,11,12,13,14,15);
  v16h A1 = __builtin_shufflevector(a10, a11, 0,1,2,3,4,5,6,7,8,9,10,11,12,13,14,15);
  v16h B0 = __builtin_shufflevector(b00, b01, 0,1,2,3,4,5,6,7,8,9,10,11,12,13,14,15);
  v16h B1 = __builtin_shufflevector(b10, b11, 0,1,2,3,4,5,6,7,8,9,10,11,12,13,14,15);

  v8f c0 = {};
  c0 = __builtin_amdgcn_wmma_f32_16x16x32_f16(false, A0, false, B0,
                                              (short)0, c0, false, false);
  c0 = __builtin_amdgcn_wmma_f32_16x16x32_f16(false, A1, false, B1,
                                              (short)0, c0, false, false);

  if (t1 < S) {                                  // scalar branch: EXEC all-1s
    v8h d00 = *(const v8h*)(kp1 +  0);
    v8h d01 = *(const v8h*)(kp1 + 16);
    v8h d10 = *(const v8h*)(kp1 + 32);
    v8h d11 = *(const v8h*)(kp1 + 48);
    v16h C0 = __builtin_shufflevector(d00, d01, 0,1,2,3,4,5,6,7,8,9,10,11,12,13,14,15);
    v16h C1 = __builtin_shufflevector(d10, d11, 0,1,2,3,4,5,6,7,8,9,10,11,12,13,14,15);

    v8f c1 = {};
    c1 = __builtin_amdgcn_wmma_f32_16x16x32_f16(false, A0, false, C0,
                                                (short)0, c1, false, false);
    c1 = __builtin_amdgcn_wmma_f32_16x16x32_f16(false, A1, false, C1,
                                                (short)0, c1, false, false);
    store_tile(out, obase, S, s0, t1, lane, c1);
  }
  store_tile(out, obase, S, s0, t0, lane, c0);
}

// ---------------------------------------------------------------------------
extern "C" void kernel_launch(void* const* d_in, const int* in_sizes, int n_in,
                              void* d_out, int out_size, void* d_ws, size_t ws_size,
                              hipStream_t stream) {
  const _Float16* mixed  = (const _Float16*)d_in[0];
  const int*      seqlen = (const int*)d_in[1];
  // d_in[2] is the python scalar batch (==64); grid must be host-static anyway.

  long long* tok_offs = (long long*)d_ws;
  long long* out_offs = tok_offs + (BATCH_N + 1);

  bmm1_prefix_kernel<<<1, 1, 0, stream>>>(seqlen, tok_offs, out_offs, BATCH_N);

  // Each wave covers a 16x32 strip; 4 waves/block -> block covers 16x128.
  // grid: x = 512/128 = 4 strips, y = 32 s-tiles, z = batch*heads.
  dim3 grid(MAXSEQ / 128, MAXSEQ / 16, BATCH_N * HEADS);
  bmm1_wmma_kernel<<<grid, 128, 0, stream>>>(mixed, seqlen, tok_offs, out_offs,
                                             (float*)d_out);
}
